// TransformerDecoder_38233798869193
// MI455X (gfx1250) — matrix operand order repaired and checked by reading, not compiled, not run on hardware
//
#include <hip/hip_runtime.h>

typedef __attribute__((ext_vector_type(16))) _Float16 v16h;
typedef __attribute__((ext_vector_type(8)))  float    v8f;
typedef int v4i __attribute__((vector_size(16)));

#define D_MODEL  512
#define NHEAD    8
#define DHEAD    64
#define DFF_N    2048
#define BATCH    8
#define SEQ      512
#define MROWS    4096            // BATCH*SEQ
#define LN_EPS   1e-5f
#define SM_SCALE 0.125f          // 1/sqrt(64)

#if defined(__HIP_DEVICE_COMPILE__) && \
    __has_builtin(__builtin_amdgcn_global_load_async_to_lds_b128) && \
    __has_builtin(__builtin_amdgcn_s_wait_asynccnt)
#define HAVE_ASYNC_LDS 1
#else
#define HAVE_ASYNC_LDS 0
#endif

#if HAVE_ASYNC_LDS
__device__ __forceinline__ __attribute__((address_space(1))) v4i*
to_gas(const void* p) { return (__attribute__((address_space(1))) v4i*)p; }
__device__ __forceinline__ __attribute__((address_space(3))) v4i*
to_las(void* p) { return (__attribute__((address_space(3))) v4i*)p; }
#endif

// A-matrix (16x32 f16) per-lane half index -> K index (ISA 7.12.2)
__device__ __forceinline__ int a_kpat(int i, int lane) {
  int p = i >> 1, w = i & 1;
  return ((p < 4) ? 0 : 16) + ((lane < 16) ? 0 : 8) + ((p & 3) << 1) + w;
}

// ---------------------------------------------------------------------------
// GEMM: C[M,N] = A[M,K] * W[N,K]^T + bias, optional ReLU.  M%64==0, N%128==0,
// K%32==0.  256 threads = 8 waves; block tile 64x128; each wave: 16x64.
// Async double-buffered LDS staging when the toolchain exposes
// GLOBAL_LOAD_ASYNC_TO_LDS (ASYNCcnt path), else direct staging.
// ---------------------------------------------------------------------------
#define BM 64
#define BN 128
#define BK 32
#define LSTR 40   // LDS row stride in halfs (80B -> every 16-half run 16B aligned)

__global__ __launch_bounds__(256) void gemm_bias_kernel(
    const float* __restrict__ A, const float* __restrict__ W,
    const float* __restrict__ bias, float* __restrict__ C,
    int M, int N, int K, int relu)
{
  __shared__ __align__(16) _Float16 As[BM][LSTR];
  __shared__ __align__(16) _Float16 Bs[BN][LSTR];
#if HAVE_ASYNC_LDS
  __shared__ __align__(16) float Araw[2][BM][BK];
  __shared__ __align__(16) float Braw[2][BN][BK];
#endif
  const int tid   = threadIdx.x;
  const int lane  = tid & 31;
  const int wv    = tid >> 5;
  const int wm    = wv >> 1;            // 0..3 : 16-row slice
  const int wn    = wv & 1;             // 0..1 : 64-col slice
  const int nlane = lane & 15;
  const int kb    = (lane < 16) ? 0 : 16;
  const int row0  = blockIdx.y * BM;
  const int col0  = blockIdx.x * BN;

  v8f zero = {};
  v8f acc[4] = {zero, zero, zero, zero};

#if HAVE_ASYNC_LDS
  // ---- async pipeline: copy raw f32 tiles with the async-LDS engine,
  // ---- convert once per tile into the f16 fragment buffers.
  #define GEMM_ISSUE_ASYNC(k0_, buf_)                                         \
    do {                                                                      \
      _Pragma("unroll")                                                       \
      for (int it = 0; it < 2; ++it) {                                        \
        int li = tid + it * 256;                                              \
        int r = li >> 3, c4 = (li & 7) << 2;                                  \
        __builtin_amdgcn_global_load_async_to_lds_b128(                       \
            to_gas(A + (size_t)(row0 + r) * K + (k0_) + c4),                  \
            to_las(&Araw[(buf_)][r][c4]), 0, 0);                              \
      }                                                                       \
      _Pragma("unroll")                                                       \
      for (int it = 0; it < 4; ++it) {                                        \
        int li = tid + it * 256;                                              \
        int r = li >> 3, c4 = (li & 7) << 2;                                  \
        __builtin_amdgcn_global_load_async_to_lds_b128(                       \
            to_gas(W + (size_t)(col0 + r) * K + (k0_) + c4),                  \
            to_las(&Braw[(buf_)][r][c4]), 0, 0);                              \
      }                                                                       \
    } while (0)

  #define GEMM_CONVERT(buf_)                                                  \
    do {                                                                      \
      _Pragma("unroll")                                                       \
      for (int it = 0; it < 2; ++it) {                                        \
        int li = tid + it * 256;                                              \
        int r = li >> 3, c4 = (li & 7) << 2;                                  \
        const float4 vq = *(const float4*)&Araw[(buf_)][r][c4];               \
        As[r][c4 + 0] = (_Float16)vq.x;  As[r][c4 + 1] = (_Float16)vq.y;      \
        As[r][c4 + 2] = (_Float16)vq.z;  As[r][c4 + 3] = (_Float16)vq.w;      \
      }                                                                       \
      _Pragma("unroll")                                                       \
      for (int it = 0; it < 4; ++it) {                                        \
        int li = tid + it * 256;                                              \
        int r = li >> 3, c4 = (li & 7) << 2;                                  \
        const float4 vq = *(const float4*)&Braw[(buf_)][r][c4];               \
        Bs[r][c4 + 0] = (_Float16)vq.x;  Bs[r][c4 + 1] = (_Float16)vq.y;      \
        Bs[r][c4 + 2] = (_Float16)vq.z;  Bs[r][c4 + 3] = (_Float16)vq.w;      \
      }                                                                       \
    } while (0)

  // prologue: tile 0
  GEMM_ISSUE_ASYNC(0, 0);
  __builtin_amdgcn_s_wait_asynccnt(0);   // own copies done; same-thread convert
  GEMM_CONVERT(0);
  __syncthreads();

  int buf = 0;
  for (int k0 = 0; k0 < K; k0 += BK) {
    const bool more = (k0 + BK) < K;
    if (more) GEMM_ISSUE_ASYNC(k0 + BK, buf ^ 1);   // overlap with WMMAs below
#else
  for (int k0 = 0; k0 < K; k0 += BK) {
    __syncthreads();
    if (k0 + BK < K) {
      __builtin_prefetch(A + (size_t)(row0 + (tid & 63))  * K + k0 + BK, 0, 0);
      __builtin_prefetch(W + (size_t)(col0 + (tid & 127)) * K + k0 + BK, 0, 0);
    }
    #pragma unroll
    for (int it = 0; it < 2; ++it) {            // 64 rows x 8 float4
      int li = tid + it * 256;
      int r = li >> 3, c4 = (li & 7) << 2;
      const float4 vq = *(const float4*)(A + (size_t)(row0 + r) * K + k0 + c4);
      As[r][c4 + 0] = (_Float16)vq.x;  As[r][c4 + 1] = (_Float16)vq.y;
      As[r][c4 + 2] = (_Float16)vq.z;  As[r][c4 + 3] = (_Float16)vq.w;
    }
    #pragma unroll
    for (int it = 0; it < 4; ++it) {            // 128 rows x 8 float4
      int li = tid + it * 256;
      int r = li >> 3, c4 = (li & 7) << 2;
      const float4 vq = *(const float4*)(W + (size_t)(col0 + r) * K + k0 + c4);
      Bs[r][c4 + 0] = (_Float16)vq.x;  Bs[r][c4 + 1] = (_Float16)vq.y;
      Bs[r][c4 + 2] = (_Float16)vq.z;  Bs[r][c4 + 3] = (_Float16)vq.w;
    }
    __syncthreads();
#endif

    // ---- fragments: preload everything, then issue 4 WMMAs back-to-back
    v16h af;
    const int mrow = wm * 16 + nlane;
    #pragma unroll
    for (int i = 0; i < 16; ++i) af[i] = As[mrow][a_kpat(i, lane)];
    v16h bfs[4];
    #pragma unroll
    for (int nt = 0; nt < 4; ++nt) {
      const int nrow = wn * 64 + nt * 16 + nlane;
      #pragma unroll
      for (int i = 0; i < 16; ++i) bfs[nt][i] = Bs[nrow][a_kpat(i, lane)];
    }
    #pragma unroll
    for (int nt = 0; nt < 4; ++nt)
      acc[nt] = __builtin_amdgcn_wmma_f32_16x16x32_f16(
          false, af, false, bfs[nt], (short)0, acc[nt], false, false);

#if HAVE_ASYNC_LDS
    if (more) {
      __syncthreads();                         // all waves done reading As/Bs
      __builtin_amdgcn_s_wait_asynccnt(0);     // next raw tile landed
      GEMM_CONVERT(buf ^ 1);
      __syncthreads();
      buf ^= 1;
    }
#endif
  }

  const int mbase = row0 + wm * 16 + ((lane < 16) ? 0 : 8);
  #pragma unroll
  for (int nt = 0; nt < 4; ++nt) {
    const int col = col0 + wn * 64 + nt * 16 + nlane;
    const float bv = bias[col];
    #pragma unroll
    for (int r = 0; r < 8; ++r) {
      float vv = acc[nt][r] + bv;
      if (relu) vv = fmaxf(vv, 0.0f);
      C[(size_t)(mbase + r) * N + col] = vv;
    }
  }
  (void)M;
}

// ---------------------------------------------------------------------------
// Flash attention over one (b,h,64-row s-tile).  Q/K/V are [B,S,512] with head
// h at columns [h*64, h*64+64).  Output written with the reference's buggy
// merge: o[b,h,s,dh] -> flat f=h*32768+s*64+dh; out[b, f&511, f>>9].
// 128 threads = 4 waves; wave w owns query rows [w*16, w*16+16).
// ---------------------------------------------------------------------------
#define KSTR 72

__global__ __launch_bounds__(128) void attention_kernel(
    const float* __restrict__ Q, const float* __restrict__ K,
    const float* __restrict__ V, float* __restrict__ Out,
    int Tlen, int causal)
{
  __shared__ __align__(16) _Float16 Qs[64][KSTR];
  __shared__ __align__(16) _Float16 Ks[64][KSTR];
  __shared__ __align__(16) _Float16 Vs[64][KSTR];
  __shared__ __align__(16) _Float16 Ps[4][16][KSTR];

  const int tid   = threadIdx.x;
  const int lane  = tid & 31;
  const int wv    = tid >> 5;            // 0..3
  const int nlane = lane & 15;
  const int kb    = (lane < 16) ? 0 : 16;
  const int rowoff= (lane < 16) ? 0 : 8;
  const int stile = blockIdx.x;          // 0..7
  const int bh    = blockIdx.y;          // 0..63
  const int b     = bh >> 3, h = bh & 7;

  const float* Qp = Q + (size_t)b * SEQ * D_MODEL + h * DHEAD;
  const float* Kp = K + (size_t)b * SEQ * D_MODEL + h * DHEAD;
  const float* Vp = V + (size_t)b * SEQ * D_MODEL + h * DHEAD;
  const int s0 = stile * 64;

  // stage Q tile 64x64 -> f16 LDS (1024 float4, 8 per thread)
  #pragma unroll
  for (int it = 0; it < 8; ++it) {
    int li = tid + it * 128;
    int r = li >> 4, c4 = (li & 15) << 2;
    const float4 vq = *(const float4*)(Qp + (size_t)(s0 + r) * D_MODEL + c4);
    Qs[r][c4 + 0] = (_Float16)vq.x;  Qs[r][c4 + 1] = (_Float16)vq.y;
    Qs[r][c4 + 2] = (_Float16)vq.z;  Qs[r][c4 + 3] = (_Float16)vq.w;
  }
  __syncthreads();

  v16h aq[2];
  {
    const int mrow = wv * 16 + nlane;
    #pragma unroll
    for (int ks = 0; ks < 2; ++ks)
      #pragma unroll
      for (int i = 0; i < 16; ++i)
        aq[ks][i] = Qs[mrow][ks * 32 + a_kpat(i, lane)];
  }

  v8f zero = {};
  v8f accO[4] = {zero, zero, zero, zero};
  float m_run[8], l_run[8];
  #pragma unroll
  for (int r = 0; r < 8; ++r) { m_run[r] = -__builtin_inff(); l_run[r] = 0.0f; }

  const int srow_base = s0 + wv * 16 + rowoff;          // + r
  const int nchunks = causal ? (stile + 1) : (Tlen >> 6);

  for (int c = 0; c < nchunks; ++c) {
    __syncthreads();
    const int t0 = c * 64;
    #pragma unroll
    for (int it = 0; it < 8; ++it) {
      int li = tid + it * 128;
      int r = li >> 4, c4 = (li & 15) << 2;
      const float4 kq = *(const float4*)(Kp + (size_t)(t0 + r) * D_MODEL + c4);
      Ks[r][c4 + 0] = (_Float16)kq.x;  Ks[r][c4 + 1] = (_Float16)kq.y;
      Ks[r][c4 + 2] = (_Float16)kq.z;  Ks[r][c4 + 3] = (_Float16)kq.w;
      const float4 vq = *(const float4*)(Vp + (size_t)(t0 + r) * D_MODEL + c4);
      Vs[r][c4 + 0] = (_Float16)vq.x;  Vs[r][c4 + 1] = (_Float16)vq.y;
      Vs[r][c4 + 2] = (_Float16)vq.z;  Vs[r][c4 + 3] = (_Float16)vq.w;
    }
    __syncthreads();

    // scores: 16x64 per wave = 4 n-tiles, K-dim = DHEAD = 2 WMMA steps
    v8f sacc[4] = {zero, zero, zero, zero};
    #pragma unroll
    for (int ks = 0; ks < 2; ++ks) {
      v16h bfs[4];
      #pragma unroll
      for (int nt = 0; nt < 4; ++nt)
        #pragma unroll
        for (int i = 0; i < 16; ++i)
          bfs[nt][i] = Ks[nt * 16 + nlane][ks * 32 + a_kpat(i, lane)];
      #pragma unroll
      for (int nt = 0; nt < 4; ++nt)
        sacc[nt] = __builtin_amdgcn_wmma_f32_16x16x32_f16(
            false, aq[ks], false, bfs[nt], (short)0, sacc[nt], false, false);
    }

    // online softmax (rows per lane: r + rowoff, col per lane: nlane)
    float sv[4][8], cmax[8];
    #pragma unroll
    for (int r = 0; r < 8; ++r) cmax[r] = -__builtin_inff();
    #pragma unroll
    for (int nt = 0; nt < 4; ++nt) {
      const int tcol = t0 + nt * 16 + nlane;
      #pragma unroll
      for (int r = 0; r < 8; ++r) {
        float x = sacc[nt][r] * SM_SCALE;
        if (causal && tcol > srow_base + r) x = -__builtin_inff();
        sv[nt][r] = x;
        cmax[r] = fmaxf(cmax[r], x);
      }
    }
    #pragma unroll
    for (int off = 1; off < 16; off <<= 1)
      #pragma unroll
      for (int r = 0; r < 8; ++r)
        cmax[r] = fmaxf(cmax[r], __shfl_xor(cmax[r], off, 32));

    float alpha[8], lsum[8];
    #pragma unroll
    for (int r = 0; r < 8; ++r) {
      float mn = fmaxf(m_run[r], cmax[r]);
      alpha[r] = __expf(m_run[r] - mn);
      m_run[r] = mn;
      lsum[r] = 0.0f;
    }
    float pv[4][8];
    #pragma unroll
    for (int nt = 0; nt < 4; ++nt)
      #pragma unroll
      for (int r = 0; r < 8; ++r) {
        float p = __expf(sv[nt][r] - m_run[r]);
        pv[nt][r] = p;
        lsum[r] += p;
      }
    #pragma unroll
    for (int off = 1; off < 16; off <<= 1)
      #pragma unroll
      for (int r = 0; r < 8; ++r)
        lsum[r] += __shfl_xor(lsum[r], off, 32);
    #pragma unroll
    for (int r = 0; r < 8; ++r) l_run[r] = l_run[r] * alpha[r] + lsum[r];
    #pragma unroll
    for (int nt = 0; nt < 4; ++nt)
      #pragma unroll
      for (int r = 0; r < 8; ++r) accO[nt][r] *= alpha[r];

    // re-fragment P via per-wave LDS (DS ops from one wave are in-order)
    #pragma unroll
    for (int nt = 0; nt < 4; ++nt)
      #pragma unroll
      for (int r = 0; r < 8; ++r)
        Ps[wv][rowoff + r][nt * 16 + nlane] = (_Float16)pv[nt][r];
    asm volatile("" ::: "memory");

    // O += P(16x64) * V(64x64): t-dim = 2 WMMA steps, 4 dh n-tiles
    #pragma unroll
    for (int ks2 = 0; ks2 < 2; ++ks2) {
      v16h ap;
      #pragma unroll
      for (int i = 0; i < 16; ++i)
        ap[i] = Ps[wv][nlane][ks2 * 32 + a_kpat(i, lane)];
      v16h bfs[4];
      #pragma unroll
      for (int nt2 = 0; nt2 < 4; ++nt2)
        #pragma unroll
        for (int i = 0; i < 16; ++i)
          bfs[nt2][i] = Vs[ks2 * 32 + a_kpat(i, lane)][nt2 * 16 + nlane];
      #pragma unroll
      for (int nt2 = 0; nt2 < 4; ++nt2)
        accO[nt2] = __builtin_amdgcn_wmma_f32_16x16x32_f16(
            false, ap, false, bfs[nt2], (short)0, accO[nt2], false, false);
    }
  }

  // epilogue with the reference's buggy merge scatter
  float* Outb = Out + (size_t)b * SEQ * D_MODEL;
  #pragma unroll
  for (int nt2 = 0; nt2 < 4; ++nt2) {
    const int dh = nt2 * 16 + nlane;
    #pragma unroll
    for (int r = 0; r < 8; ++r) {
      const int s = srow_base + r;
      const float o = accO[nt2][r] / l_run[r];
      const int f = h * (SEQ * DHEAD) + s * DHEAD + dh;
      Outb[(size_t)(f & 511) * D_MODEL + (f >> 9)] = o;
    }
  }
}

// ---------------------------------------------------------------------------
// out[row] = LayerNorm(A[row] + Bres[row]) * gamma + beta  (row length 512)
// 128 threads, 1 float4 each.
// ---------------------------------------------------------------------------
__global__ __launch_bounds__(128) void ln_residual_kernel(
    const float* __restrict__ A, const float* __restrict__ Bres,
    const float* __restrict__ gamma, const float* __restrict__ beta,
    float* __restrict__ Out)
{
  __shared__ float red[2][4];
  const int row = blockIdx.x;
  const int tid = threadIdx.x;
  const int c = tid << 2;
  const float4 av = *(const float4*)(A    + (size_t)row * D_MODEL + c);
  const float4 bv = *(const float4*)(Bres + (size_t)row * D_MODEL + c);
  const float x0 = av.x + bv.x, x1 = av.y + bv.y, x2 = av.z + bv.z, x3 = av.w + bv.w;
  float s  = x0 + x1 + x2 + x3;
  float sq = x0 * x0 + x1 * x1 + x2 * x2 + x3 * x3;
  #pragma unroll
  for (int off = 1; off < 32; off <<= 1) {
    s  += __shfl_xor(s,  off, 32);
    sq += __shfl_xor(sq, off, 32);
  }
  if ((tid & 31) == 0) { red[0][tid >> 5] = s; red[1][tid >> 5] = sq; }
  __syncthreads();
  const float ts = red[0][0] + red[0][1] + red[0][2] + red[0][3];
  const float tq = red[1][0] + red[1][1] + red[1][2] + red[1][3];
  const float mean = ts * (1.0f / D_MODEL);
  const float var  = tq * (1.0f / D_MODEL) - mean * mean;
  const float inv  = rsqrtf(var + LN_EPS);
  const float4 gv = *(const float4*)(gamma + c);
  const float4 tb = *(const float4*)(beta  + c);
  float4 ov;
  ov.x = (x0 - mean) * inv * gv.x + tb.x;
  ov.y = (x1 - mean) * inv * gv.y + tb.y;
  ov.z = (x2 - mean) * inv * gv.z + tb.z;
  ov.w = (x3 - mean) * inv * gv.w + tb.w;
  *(float4*)(Out + (size_t)row * D_MODEL + c) = ov;
}

// ---------------------------------------------------------------------------
extern "C" void kernel_launch(void* const* d_in, const int* in_sizes, int n_in,
                              void* d_out, int out_size, void* d_ws, size_t ws_size,
                              hipStream_t stream) {
  const float* x     = (const float*)d_in[0];
  const float* cross = (const float*)d_in[1];
  // att1
  const float *wq1 = (const float*)d_in[2],  *bq1 = (const float*)d_in[3];
  const float *wk1 = (const float*)d_in[4],  *bk1 = (const float*)d_in[5];
  const float *wv1 = (const float*)d_in[6],  *bv1 = (const float*)d_in[7];
  const float *wo1 = (const float*)d_in[8],  *bo1 = (const float*)d_in[9];
  // att2
  const float *wq2 = (const float*)d_in[10], *bq2 = (const float*)d_in[11];
  const float *wk2 = (const float*)d_in[12], *bk2 = (const float*)d_in[13];
  const float *wv2 = (const float*)d_in[14], *bv2 = (const float*)d_in[15];
  const float *wo2 = (const float*)d_in[16], *bo2 = (const float*)d_in[17];
  // ffn + norms
  const float *fw1 = (const float*)d_in[18], *fb1 = (const float*)d_in[19];
  const float *fw2 = (const float*)d_in[20], *fb2 = (const float*)d_in[21];
  const float *g1 = (const float*)d_in[22], *be1 = (const float*)d_in[23];
  const float *g2 = (const float*)d_in[24], *be2 = (const float*)d_in[25];
  const float *g3 = (const float*)d_in[26], *be3 = (const float*)d_in[27];

  float* ws = (float*)d_ws;
  const size_t SLOT = (size_t)MROWS * D_MODEL;   // 2,097,152 floats = 8 MB
  float* q   = ws + 0 * SLOT;
  float* k   = ws + 1 * SLOT;
  float* v   = ws + 2 * SLOT;
  float* mg  = ws + 3 * SLOT;
  float* t1  = ws + 4 * SLOT;
  float* xn1 = ws + 5 * SLOT;
  float* xn2 = ws + 6 * SLOT;
  float* ffh = ws + 0 * SLOT;   // reuses q/k/v/mg (free during FFN): 4096x2048
  float* out = (float*)d_out;

  const dim3 gP(D_MODEL / BN, MROWS / BM);   // (4, 64)  N=512 GEMMs
  const dim3 gF1(DFF_N / BN, MROWS / BM);    // (16, 64) N=2048 GEMM
  const dim3 gAtt(SEQ / 64, BATCH * NHEAD);  // (8, 64)
  const dim3 bG(256), bA(128), bL(128);

  // ---- self attention ----
  gemm_bias_kernel<<<gP, bG, 0, stream>>>(x, wq1, bq1, q, MROWS, D_MODEL, D_MODEL, 0);
  gemm_bias_kernel<<<gP, bG, 0, stream>>>(x, wk1, bk1, k, MROWS, D_MODEL, D_MODEL, 0);
  gemm_bias_kernel<<<gP, bG, 0, stream>>>(x, wv1, bv1, v, MROWS, D_MODEL, D_MODEL, 0);
  attention_kernel<<<gAtt, bA, 0, stream>>>(q, k, v, mg, SEQ, 1);
  gemm_bias_kernel<<<gP, bG, 0, stream>>>(mg, wo1, bo1, t1, MROWS, D_MODEL, D_MODEL, 0);
  ln_residual_kernel<<<MROWS, bL, 0, stream>>>(t1, x, g1, be1, xn1);

  // ---- cross attention ----
  gemm_bias_kernel<<<gP, bG, 0, stream>>>(xn1,   wq2, bq2, q, MROWS, D_MODEL, D_MODEL, 0);
  gemm_bias_kernel<<<gP, bG, 0, stream>>>(cross, wk2, bk2, k, MROWS, D_MODEL, D_MODEL, 0);
  gemm_bias_kernel<<<gP, bG, 0, stream>>>(cross, wv2, bv2, v, MROWS, D_MODEL, D_MODEL, 0);
  attention_kernel<<<gAtt, bA, 0, stream>>>(q, k, v, mg, SEQ, 0);
  gemm_bias_kernel<<<gP, bG, 0, stream>>>(mg, wo2, bo2, t1, MROWS, D_MODEL, D_MODEL, 0);
  ln_residual_kernel<<<MROWS, bL, 0, stream>>>(t1, xn1, g2, be2, xn2);

  // ---- FFN ----
  gemm_bias_kernel<<<gF1, bG, 0, stream>>>(xn2, fw1, fb1, ffh, MROWS, DFF_N, D_MODEL, 1);
  gemm_bias_kernel<<<gP,  bG, 0, stream>>>(ffh, fw2, fb2, t1,  MROWS, D_MODEL, DFF_N, 0);
  ln_residual_kernel<<<MROWS, bL, 0, stream>>>(t1, xn2, g3, be3, out);

  (void)in_sizes; (void)n_in; (void)out_size; (void)ws_size;
}
